// TensorProductBlock_54589034332836
// MI455X (gfx1250) — matrix operand order repaired and checked by reading, not compiled, not run on hardware
//
#include <hip/hip_runtime.h>

typedef __attribute__((ext_vector_type(16))) __bf16 v16bf;
typedef __attribute__((ext_vector_type(8)))  float  v8f;

#define SV5_C 52
#define NSP   32768   // 32*32*32

// ---------------- init: zero the atomicMax accumulators ----------------
__global__ void k_init(unsigned* maxima) {
  if (threadIdx.x < 64) maxima[threadIdx.x] = 0u;
}

// ---------------- scalar-channel sums (deterministic two-pass) ----------------
__global__ __launch_bounds__(256) void k_sum(const float* __restrict__ sv5,
                                             float* __restrict__ partials) {
  int b = blockIdx.x >> 7;
  int s = ((blockIdx.x & 127) << 8) | threadIdx.x;
  const float* base = sv5 + (size_t)b * SV5_C * NSP + s;
  __shared__ float red[8][8];
  float v[8];
  for (int m = 0; m < 8; ++m) v[m] = base[m * NSP];
  for (int m = 0; m < 8; ++m)
    for (int off = 16; off > 0; off >>= 1) v[m] += __shfl_xor(v[m], off, 32);
  int wid = threadIdx.x >> 5, lane = threadIdx.x & 31;
  if (lane == 0)
    for (int m = 0; m < 8; ++m) red[wid][m] = v[m];
  __syncthreads();
  if (threadIdx.x < 8) {
    float t = 0.f;
    for (int wv = 0; wv < 8; ++wv) t += red[wv][threadIdx.x];
    partials[blockIdx.x * 8 + threadIdx.x] = t;
  }
}

__global__ void k_mean(const float* __restrict__ partials, float* __restrict__ mean) {
  int m = threadIdx.x;
  if (m < 8) {
    float t = 0.f;
    for (int i = 0; i < 256; ++i) t += partials[i * 8 + m];
    mean[m] = t * (1.f / 65536.f);
  }
}

// ---------------- per-field max of squared norm over space, per batch ----------------
__global__ __launch_bounds__(256) void k_stats(const float* __restrict__ sv5,
                                               const float* __restrict__ mean,
                                               unsigned* __restrict__ maxima) {
  int b = blockIdx.x >> 7;
  int s = ((blockIdx.x & 127) << 8) | threadIdx.x;
  const float* base = sv5 + (size_t)b * SV5_C * NSP + s;
  float vals[20];
  for (int m = 0; m < 8; ++m) { float x = base[m * NSP] - mean[m]; vals[m] = x * x; }
  for (int m = 0; m < 8; ++m) {
    float a = base[(8 + 3 * m + 0) * NSP];
    float c = base[(8 + 3 * m + 1) * NSP];
    float d = base[(8 + 3 * m + 2) * NSP];
    vals[8 + m] = a * a + c * c + d * d;           // |v|^2 for l=1 block
  }
  for (int m = 0; m < 4; ++m) {
    float t = 0.f;
    for (int k = 0; k < 5; ++k) { float x = base[(32 + 5 * m + k) * NSP]; t += x * x; }
    vals[16 + m] = t;
  }
  int lane = threadIdx.x & 31;
  for (int i = 0; i < 20; ++i) {
    float v = vals[i];
    for (int off = 16; off > 0; off >>= 1) v = fmaxf(v, __shfl_xor(v, off, 32));
    if (lane == 0) {
      int slot = (i < 8) ? i : (i < 16 ? 8 + (i - 8) : 16 + (i - 16));
      atomicMax(&maxima[slot * 2 + b], __float_as_uint(v));
      if (i >= 8 && i < 16)   // tensor-product block: norm^2 = |v|^4
        atomicMax(&maxima[(24 + (i - 8)) * 2 + b], __float_as_uint(v * v));
    }
  }
}

__global__ void k_scale(const unsigned* __restrict__ maxima,
                        const float* __restrict__ bn_scale,
                        float* __restrict__ scale) {
  int t = threadIdx.x;
  if (t < 32) {
    float m0 = __uint_as_float(maxima[t * 2 + 0]);
    float m1 = __uint_as_float(maxima[t * 2 + 1]);
    float stat = 0.5f * (m0 + m1);
    scale[t] = rsqrtf(stat + 1e-5f) * bn_scale[t];
  }
}

// ---------------- expand + normalize + pad -> channels-last bf16 ----------------
__global__ __launch_bounds__(128) void k_build_xp(const float* __restrict__ sv5,
                                                  const float* __restrict__ mean,
                                                  const float* __restrict__ scale,
                                                  __bf16* __restrict__ Xp) {
  int vox = blockIdx.x;                 // [b][zz][yy][xx] over 2*36^3
  int b = vox / 46656; int r = vox % 46656;
  int zz = r / 1296; int yy = (r / 36) % 36; int xx = r % 36;
  int c = threadIdx.x;
  float val = 0.f;
  bool interior = (zz >= 2 && zz < 34 && yy >= 2 && yy < 34 && xx >= 2 && xx < 34);
  if (interior && c < 124) {
    int s = ((zz - 2) * 32 + (yy - 2)) * 32 + (xx - 2);
    const float* base = sv5 + (size_t)b * SV5_C * NSP + s;
    if (c < 8)        val = (base[c * NSP] - mean[c]) * scale[c];
    else if (c < 32) { int q = c - 8;  val = base[(8 + q) * NSP]  * scale[8  + q / 3]; }
    else if (c < 52) { int q = c - 32; val = base[(32 + q) * NSP] * scale[16 + q / 5]; }
    else {
      int q = c - 52; int m = q / 9; int i = (q / 3) % 3; int j = q % 3;
      val = base[(8 + 3 * m + i) * NSP] * base[(8 + 3 * m + j) * NSP] * scale[24 + m];
    }
  }
  Xp[(size_t)vox * 128 + c] = (__bf16)val;
}

// ---------------- assemble conv kernel -> bf16 [tap][co=64][ci=128] ----------------
__global__ __launch_bounds__(256) void k_build_w(const float* __restrict__ w,
                                                 const float* __restrict__ basis,
                                                 __bf16* __restrict__ Wl) {
  int t = blockIdx.x * 256 + threadIdx.x;
  if (t >= 125 * 64 * 128) return;
  int tap = t >> 13;
  int r = t & 8191;
  int co = r >> 7;
  int ci = r & 127;
  float val = 0.f;
  if (co < 52 && ci < 124) {
    int i, u, p, do_;
    if (co < 8)       { i = 0; u = co;            p = 0;             do_ = 1; }
    else if (co < 32) { i = 1; u = (co - 8) / 3;  p = (co - 8) % 3;  do_ = 3; }
    else              { i = 2; u = (co - 32) / 5; p = (co - 32) % 5; do_ = 5; }
    int j, v, q, di;
    if (ci < 8)       { j = 0; v = ci;            q = 0;             di = 1; }
    else if (ci < 32) { j = 1; v = (ci - 8) / 3;  q = (ci - 8) % 3;  di = 3; }
    else if (ci < 52) { j = 2; v = (ci - 32) / 5; q = (ci - 32) % 5; di = 5; }
    else              { j = 3; v = (ci - 52) / 9; q = (ci - 52) % 9; di = 9; }
    int K = 3 * (do_ < di ? do_ : di);
    const float* wp = w + ((((size_t)i * 4 + j) * 8 + u) * 8 + v) * 15;
    const float* bp = basis + ((size_t)(i * 4 + j) * 15) * 5625
                            + (size_t)(p * 9 + q) * 125 + tap;
    for (int kb = 0; kb < K; ++kb) val += wp[kb] * bp[(size_t)kb * 5625];
  }
  Wl[t] = (__bf16)val;
}

// ---------------- implicit-GEMM conv via v_wmma_f32_16x16x32_bf16 ----------------
// 128 threads = 4 waves; each wave owns one 16-co block and BOTH 16-x blocks
// (2 accumulators). A fragment loaded once per K-step and reused for 2 WMMAs:
// per-wmma operand traffic = 1 global b128-pair + 2 ds b128 (vs 2+2 before).
__global__ __launch_bounds__(128) void k_conv(const __bf16* __restrict__ Xp,
                                              const __bf16* __restrict__ Wl,
                                              float* __restrict__ out) {
  __shared__ uint4 smem[576];                 // one (b,zz,yy) row: 36 x * 128 ci bf16
  const int y = blockIdx.x, z = blockIdx.y, b = blockIdx.z;
  const int tid = threadIdx.x;
  const int lane = tid & 31, wave = tid >> 5;
  const int lhi = lane >> 4, lmod = lane & 15;
  const int co_base = wave << 4;              // 4 co blocks of 16
  v8f acc0 = {0.f, 0.f, 0.f, 0.f, 0.f, 0.f, 0.f, 0.f};
  v8f acc1 = {0.f, 0.f, 0.f, 0.f, 0.f, 0.f, 0.f, 0.f};

  const size_t plane = ((size_t)b * 36 + z) * 36 + y;  // row(r) = plane + (r/5)*36 + r%5

  // preload row 0
  {
    const uint4* src = (const uint4*)(Xp + (plane) * 4608);
    smem[tid]       = src[tid];
    smem[tid + 128] = src[tid + 128];
    smem[tid + 256] = src[tid + 256];
    smem[tid + 384] = src[tid + 384];
    if (tid < 64) smem[tid + 512] = src[tid + 512];
  }
  __syncthreads();

  for (int row = 0; row < 25; ++row) {
    const int dz = row / 5, dy = row % 5;
    // issue next row's global loads BEFORE compute so latency hides under WMMAs
    uint4 n0 = {}, n1 = {}, n2 = {}, n3 = {}, n4 = {};
    const bool have_next = row < 24;
    if (have_next) {
      const int nr = row + 1;
      const uint4* src = (const uint4*)(Xp + (plane + (nr / 5) * 36 + nr % 5) * 4608);
      n0 = src[tid];
      n1 = src[tid + 128];
      n2 = src[tid + 256];
      n3 = src[tid + 384];
      if (tid < 64) n4 = src[tid + 512];
      // prefetch two rows ahead (global_prefetch_b8), one 128B line per lane
      if (row + 2 <= 24 && tid < 72) {
        const int pr = row + 2;
        const char* p = (const char*)(Xp + (plane + (pr / 5) * 36 + pr % 5) * 4608);
        __builtin_prefetch(p + tid * 128, 0, 1);
      }
    }

    const __bf16* wrow = Wl + (size_t)((dz * 5 + dy) * 5) * 8192;
    for (int dx = 0; dx < 5; ++dx) {
      const __bf16* wtap = wrow + (size_t)dx * 8192 + (size_t)(co_base + lmod) * 128;
      const int xb = (lmod + dx) * 128;
      for (int c = 0; c < 4; ++c) {
        union { uint4 u[2]; v16bf v; } fa, fb0, fb1;
        // A fragment: k = (lane>=16?16:0)+h (contiguous ci), m = lane&15
        const int ao = c * 32 + (lhi << 4);
        fa.u[0] = *(const uint4*)(wtap + ao);
        fa.u[1] = *(const uint4*)(wtap + ao + 8);
        // B fragments: k = (lane>=16?16:0)+h (contiguous ci), n = lane&15
        const int bo = (xb + c * 32 + (lhi << 4)) >> 3;   // uint4 index, n-block 0
        fb0.u[0] = smem[bo];
        fb0.u[1] = smem[bo + 1];
        fb1.u[0] = smem[bo + 256];                        // n-block 1: +16*128 elems
        fb1.u[1] = smem[bo + 257];
        acc0 = __builtin_amdgcn_wmma_f32_16x16x32_bf16(false, fa.v, false, fb0.v,
                                                       (short)0, acc0, false, false);
        acc1 = __builtin_amdgcn_wmma_f32_16x16x32_bf16(false, fa.v, false, fb1.v,
                                                       (short)0, acc1, false, false);
      }
    }

    if (have_next) {
      __syncthreads();     // all waves done reading current row
      smem[tid]       = n0;
      smem[tid + 128] = n1;
      smem[tid + 256] = n2;
      smem[tid + 384] = n3;
      if (tid < 64) smem[tid + 512] = n4;
      __syncthreads();     // row ready
    }
  }

  // epilogue: C layout -> m = r + (lane>=16?8:0), n = lane&15
  for (int nb = 0; nb < 2; ++nb) {
    const v8f acc = nb ? acc1 : acc0;
    const int x = (nb << 4) + lmod;
    for (int r = 0; r < 8; ++r) {
      int co = co_base + r + (lhi << 3);
      if (co < 52) {
        float v = acc[r];
        if (co < 8) v = fmaxf(v, 0.f);   // ScalarActivation relu
        out[((((size_t)b * 52 + co) * 32 + z) * 32 + y) * 32 + x] = v;
      }
    }
  }
}

extern "C" void kernel_launch(void* const* d_in, const int* in_sizes, int n_in,
                              void* d_out, int out_size, void* d_ws, size_t ws_size,
                              hipStream_t stream) {
  (void)in_sizes; (void)n_in; (void)out_size; (void)ws_size;
  const float* sv5      = (const float*)d_in[0];
  const float* w        = (const float*)d_in[1];
  const float* basis    = (const float*)d_in[2];
  const float* bn_scale = (const float*)d_in[3];
  float* out = (float*)d_out;

  char* ws = (char*)d_ws;
  const size_t XP_BYTES = 2ull * 36 * 36 * 36 * 128 * 2;   // 23,887,872 B
  const size_t WL_BYTES = 125ull * 64 * 128 * 2;           //  2,048,000 B
  __bf16* Xp = (__bf16*)ws;
  __bf16* Wl = (__bf16*)(ws + XP_BYTES);
  float*    partials = (float*)(ws + XP_BYTES + WL_BYTES); // 2048 floats
  unsigned* maxima   = (unsigned*)(partials + 2048);       // 64 u32
  float*    scale    = (float*)(maxima + 64);              // 32 floats
  float*    mean     = scale + 32;                         // 8 floats

  k_init    <<<1, 64, 0, stream>>>(maxima);
  k_sum     <<<256, 256, 0, stream>>>(sv5, partials);
  k_mean    <<<1, 8, 0, stream>>>(partials, mean);
  k_stats   <<<256, 256, 0, stream>>>(sv5, mean, maxima);
  k_scale   <<<1, 32, 0, stream>>>(maxima, bn_scale, scale);
  k_build_xp<<<2 * 36 * 36 * 36, 128, 0, stream>>>(sv5, mean, scale, Xp);
  k_build_w <<<(125 * 64 * 128 + 255) / 256, 256, 0, stream>>>(w, basis, Wl);
  k_conv    <<<dim3(32, 32, 2), 128, 0, stream>>>(Xp, Wl, out);
}
